// SelfAttention_71133248356710
// MI455X (gfx1250) — matrix operand order repaired and checked by reading, not compiled, not run on hardware
//
#include <hip/hip_runtime.h>

typedef __attribute__((ext_vector_type(16))) _Float16 v16h;
typedef __attribute__((ext_vector_type(8)))  _Float16 v8h;
typedef __attribute__((ext_vector_type(8)))  float    v8f;
typedef __attribute__((ext_vector_type(4)))  unsigned int u32x4;
typedef __attribute__((ext_vector_type(8)))  int          i32x8;
typedef __attribute__((ext_vector_type(4)))  int          i32x4;

#define SEQ   2048
#define HID   4096
#define NH    32
#define HD    128
#define KVG   2
#define QKVD  (NH*HD + 2*KVG*HD)   /* 4608 */
#define QSZ   (NH*HD)              /* 4096 */

// ---------------------------------------------------------------------------
// WMMA helpers (CDNA5 16x16x32 f16 -> f32)
// ---------------------------------------------------------------------------
__device__ __forceinline__ v8f wmma16(v16h a, v16h b, v8f c) {
  return __builtin_amdgcn_wmma_f32_16x16x32_f16(false, a, false, b, (short)0, c,
                                                false, false);
}

// A-matrix 16x32 f16 fragment: lane m = lane&15; lanes 0-15 hold K in
// [0,8) U [16,24), lanes 16-31 hold K in [8,16) U [24,32).
__device__ __forceinline__ v16h frag_a(const _Float16* p, int hb) {
  const int b0 = hb * 8;
  v8h lo = *(const v8h*)(p + b0);
  v8h hi = *(const v8h*)(p + b0 + 16);
  v16h r;
#pragma unroll
  for (int i = 0; i < 8; ++i) { r[i] = lo[i]; r[i + 8] = hi[i]; }
  return r;
}

// B-matrix 32x16 f16 fragment: lane n = lane&15; lanes 0-15 hold K in
// [0,8) U [16,24), lanes 16-31 hold K in [8,16) U [24,32).  p points at (col_for_this_lane, 0), K contiguous.
__device__ __forceinline__ v16h frag_b(const _Float16* p, int hb) {
  const int b0 = hb * 8;
  v8h lo = *(const v8h*)(p + b0);
  v8h hi = *(const v8h*)(p + b0 + 16);
  v16h r;
#pragma unroll
  for (int i = 0; i < 8; ++i) { r[i] = lo[i]; r[i + 8] = hi[i]; }
  return r;
}

// ---------------------------------------------------------------------------
// Tensor Data Mover: 2D tile (td1 rows x td0 elems, f16, row stride in elems)
// from global into LDS.  Issue from ONE wave; wait with tdm_wait(); then
// __syncthreads() to release consumers.
// This toolchain's builtin is the 6-arg form:
//   (uint32x4 g0, int32x8 g1, int32x4 g2, int32x4 g3, int32x8 g4, i32 cpol)
// ---------------------------------------------------------------------------
#if __has_builtin(__builtin_amdgcn_tensor_load_to_lds)
#define HAVE_TDM 1
__device__ __forceinline__ void tdm_load_2d(_Float16* lptr, const void* gptr,
                                            unsigned td0, unsigned td1,
                                            unsigned stride) {
  unsigned lds_off = (unsigned)(size_t)lptr;
  unsigned long long ga = (unsigned long long)(size_t)gptr;
  // D# group0: count=1 | lds_addr | global_addr[56:0] | type=2
  u32x4 g0 = { 1u, lds_off, (unsigned)ga,
               (unsigned)((ga >> 32) & 0x1FFFFFFu) | 0x80000000u };
  // D# group1: data_size=2B; tensor_dim0=td0; tensor_dim1=td1; tile_dim0=td0;
  //            tile_dim1=td1; tensor_dim0_stride=stride
  i32x8 g1 = { (int)(1u << 16),          // [17:16] data_size=1 (2 bytes)
               (int)(td0 << 16),         // [63:48] tensor_dim0 lo
               (int)(td1 << 16),         // [95:80] tensor_dim1 lo
               (int)(td0 << 16),         // [127:112] tile_dim0
               (int)td1,                 // [143:128] tile_dim1 (tile_dim2=0)
               (int)stride,              // [191:160] tensor_dim0_stride lo
               0, 0 };
  i32x4 gz4 = {0, 0, 0, 0};
  i32x8 gz8 = {0, 0, 0, 0, 0, 0, 0, 0};
  __builtin_amdgcn_tensor_load_to_lds(g0, g1, gz4, gz4, gz8, 0);
}
__device__ __forceinline__ void tdm_wait() {
  __builtin_amdgcn_s_wait_tensorcnt(0);
}
#else
#define HAVE_TDM 0
// Wave-cooperative fallback (issuing wave's 32 lanes copy the tile).
__device__ __forceinline__ void tdm_load_2d(_Float16* lptr, const void* gptr,
                                            unsigned td0, unsigned td1,
                                            unsigned stride) {
  const _Float16* g = (const _Float16*)gptr;
  unsigned lane = threadIdx.x & 31;
  unsigned total = td0 * td1;
  for (unsigned idx = lane * 8; idx < total; idx += 32 * 8) {
    unsigned row = idx / td0, col = idx - row * td0;
    *(v8h*)(lptr + idx) = *(const v8h*)(g + (size_t)row * stride + col);
  }
}
__device__ __forceinline__ void tdm_wait() {}
#endif

// ---------------------------------------------------------------------------
// f32 -> f16 cast (4-wide)
// ---------------------------------------------------------------------------
__global__ __launch_bounds__(256) void cast_f16(const float* __restrict__ in,
                                                _Float16* __restrict__ out,
                                                int n4) {
  int i = blockIdx.x * blockDim.x + threadIdx.x;
  if (i < n4) {
    float4 v = ((const float4*)in)[i];
    _Float16 tmp[4] = {(_Float16)v.x, (_Float16)v.y, (_Float16)v.z, (_Float16)v.w};
    *(short4*)((short*)out + (size_t)i * 4) = *(short4*)tmp;
  }
}

// ---------------------------------------------------------------------------
// Tiled transpose + cast:  in f32 (K x N)  ->  out f16 (N x K)
// One-time pass so the GEMM can stage weights with contiguous TDM tiles.
// ---------------------------------------------------------------------------
__global__ __launch_bounds__(256) void transpose_cast(const float* __restrict__ in,
                                                      _Float16* __restrict__ out,
                                                      int K, int N) {
  __shared__ float tile[32][33];
  const int n0 = blockIdx.x * 32, k0 = blockIdx.y * 32;
  const int tx = threadIdx.x & 31, ty = threadIdx.x >> 5;  // 32 x 8
#pragma unroll
  for (int i = 0; i < 32; i += 8)
    tile[ty + i][tx] = in[(size_t)(k0 + ty + i) * N + n0 + tx];
  __syncthreads();
#pragma unroll
  for (int i = 0; i < 32; i += 8)
    out[(size_t)(n0 + ty + i) * K + k0 + tx] = (_Float16)tile[tx][ty + i];
}

// ---------------------------------------------------------------------------
// WMMA GEMM:  C(f32, MxN) = A(f16, MxK row-major) @ Bt(f16, NxK row-major)^T
// Block tile 128x128, K-step 32, TDM-staged double-buffered LDS.
// 8 waves, each computing 32x64 (2x4 WMMA tiles, 8 wmma / K-step).
// ---------------------------------------------------------------------------
__global__ __launch_bounds__(256) void gemm_f16(const _Float16* __restrict__ A,
                                                const _Float16* __restrict__ Bt,
                                                const float* __restrict__ bias,
                                                float* __restrict__ C,
                                                int M, int N, int K) {
  __shared__ __align__(16) _Float16 As[2][128 * 32];
  __shared__ __align__(16) _Float16 Bs[2][128 * 32];

  const int m0 = blockIdx.x * 128;
  const int n0 = blockIdx.y * 128;
  const int tid = threadIdx.x;
  const int lane = tid & 31;
  const int w = tid >> 5;
  const int wm = (w & 3) * 32;   // wave 32-row slice
  const int wn = (w >> 2) * 64;  // wave 64-col slice
  const int ln = lane & 15;
  const int hb = (lane >> 4) & 1;

  const int nk = K >> 5;

  // prologue: stage tile 0 into buffer 0 via TDM
  if (w == 0) {
    tdm_load_2d(&As[0][0], A + (size_t)m0 * K, 32, 128, K);
    tdm_load_2d(&Bs[0][0], Bt + (size_t)n0 * K, 32, 128, K);
    tdm_wait();
  }
  __syncthreads();

  v8f acc[2][4] = {};

  for (int i = 0; i < nk; ++i) {
    const int cur = i & 1;
    const int nxt = cur ^ 1;
    // async-stage next tile while computing on current one
    if (w == 0 && i + 1 < nk) {
      tdm_load_2d(&As[nxt][0], A + (size_t)m0 * K + (i + 1) * 32, 32, 128, K);
      tdm_load_2d(&Bs[nxt][0], Bt + (size_t)n0 * K + (i + 1) * 32, 32, 128, K);
    }

    v16h af[2], bf[4];
#pragma unroll
    for (int ii = 0; ii < 2; ++ii)
      af[ii] = frag_a(&As[cur][(wm + ii * 16 + ln) * 32], hb);
#pragma unroll
    for (int jj = 0; jj < 4; ++jj)
      bf[jj] = frag_b(&Bs[cur][(wn + jj * 16 + ln) * 32], hb);
#pragma unroll
    for (int ii = 0; ii < 2; ++ii)
#pragma unroll
      for (int jj = 0; jj < 4; ++jj)
        acc[ii][jj] = wmma16(af[ii], bf[jj], acc[ii][jj]);

    if (w == 0 && i + 1 < nk) tdm_wait();
    __syncthreads();
  }

  // epilogue: C layout, VGPR r -> row (r + 8*hb), lane&15 -> col
  const int mb = hb * 8;
#pragma unroll
  for (int ii = 0; ii < 2; ++ii) {
#pragma unroll
    for (int jj = 0; jj < 4; ++jj) {
      int col = n0 + wn + jj * 16 + ln;
      float badd = bias ? bias[col] : 0.0f;
#pragma unroll
      for (int r = 0; r < 8; ++r) {
        int row = m0 + wm + ii * 16 + mb + r;
        C[(size_t)row * N + col] = acc[ii][jj][r] + badd;
      }
    }
  }
}

// ---------------------------------------------------------------------------
// RoPE + split + relayout:
//  qkv f32 (SEQ x 4608) -> Qh f16 [NH][SEQ][HD] (rope, * 1/sqrt(HD)),
//  Kh f16 [KVG][SEQ][HD] (rope), Vt f16 [KVG][HD][SEQ] (transposed)
// ---------------------------------------------------------------------------
__global__ __launch_bounds__(128) void rope_split(const float* __restrict__ qkv,
                                                  const int* __restrict__ pos,
                                                  _Float16* __restrict__ Qh,
                                                  _Float16* __restrict__ Kh,
                                                  _Float16* __restrict__ Vt) {
  const int bid = blockIdx.x;
  const int t = bid / 36;
  const int hh = bid - t * 36;
  const int d = threadIdx.x;
  const float p = (float)pos[t];

  if (hh < NH + KVG) {  // q head or k group: apply rope
    const float* x;
    if (hh < NH) x = qkv + (size_t)t * QKVD + hh * HD;
    else         x = qkv + (size_t)t * QKVD + QSZ + (hh - NH) * HD;
    float o;
    if (d < 64) {
      int i = d >> 1;
      float theta = __powf(10000.0f, -((float)(2 * i)) / 64.0f);
      float ang = p * theta;
      float c = __cosf(ang), s = __sinf(ang);
      float x0 = x[2 * i], x1 = x[2 * i + 1];
      o = (d & 1) ? (x1 * c + x0 * s) : (x0 * c - x1 * s);
    } else {
      o = x[d];
    }
    if (hh < NH) {
      Qh[((size_t)hh * SEQ + t) * HD + d] = (_Float16)(o * 0.08838834764831845f);
    } else {
      Kh[((size_t)(hh - NH) * SEQ + t) * HD + d] = (_Float16)o;
    }
  } else {  // v group: transpose-store (d-major)
    int g = hh - (NH + KVG);
    const float* x = qkv + (size_t)t * QKVD + QSZ + KVG * HD + g * HD;
    Vt[((size_t)g * HD + d) * SEQ + t] = (_Float16)x[d];
  }
}

// ---------------------------------------------------------------------------
// Flash attention, causal.  grid (SEQ/128, NH), 8 waves x 16 query rows.
// K/V tiles TDM-staged; per 32-key chunk: 8 wmma scores + 8 wmma O += P V.
// ---------------------------------------------------------------------------
__global__ __launch_bounds__(256) void attn(const _Float16* __restrict__ Qh,
                                            const _Float16* __restrict__ Kh,
                                            const _Float16* __restrict__ Vt,
                                            _Float16* __restrict__ ctx) {
  __shared__ __align__(16) _Float16 Ks[32 * 128];     // [key][d]
  __shared__ __align__(16) _Float16 Vs[128 * 32];     // [d][key]
  __shared__ __align__(16) _Float16 Ps[8 * 16 * 32];  // per-wave P tile

  const int h = blockIdx.y;
  const int g = h >> 4;  // 16 heads per KV group
  const int Q0 = blockIdx.x * 128;
  const int tid = threadIdx.x;
  const int lane = tid & 31;
  const int w = tid >> 5;
  const int ln = lane & 15;
  const int hb = (lane >> 4) & 1;
  const int mb = hb * 8;
  const int q0 = Q0 + w * 16;

  // Q fragments: 16 rows x 128 dims (scale pre-folded into Qh)
  const _Float16* Qbase = Qh + ((size_t)h * SEQ + q0 + ln) * HD;
  v16h qf[4];
#pragma unroll
  for (int kd = 0; kd < 4; ++kd) qf[kd] = frag_a(Qbase + kd * 32, hb);

  v8f acc[8] = {};
  float mrow[8], lrow[8];
#pragma unroll
  for (int r = 0; r < 8; ++r) { mrow[r] = -3.0e38f; lrow[r] = 0.0f; }

  _Float16* Pw = Ps + w * 16 * 32;
  const int nkb = (Q0 + 128) >> 5;

  for (int kb = 0; kb < nkb; ++kb) {
    const int k32 = kb * 32;
    if (w == 0) {
      // K tile: 32 contiguous rows of 128 halves -> one linear TDM copy
      tdm_load_2d(&Ks[0], Kh + ((size_t)g * SEQ + k32) * HD, 32 * HD, 1, 32 * HD);
      // V tile: 128 rows (d) x 32 keys, row stride SEQ
      tdm_load_2d(&Vs[0], Vt + (size_t)g * HD * SEQ + k32, 32, 128, SEQ);
      tdm_wait();
    }
    __syncthreads();

    if (k32 <= q0 + 15) {  // wave-uniform causal guard
      v8f s[2] = {};
#pragma unroll
      for (int tn = 0; tn < 2; ++tn)
#pragma unroll
        for (int kd = 0; kd < 4; ++kd) {
          v16h kf = frag_b(Ks + (tn * 16 + ln) * 128 + kd * 32, hb);
          s[tn] = wmma16(qf[kd], kf, s[tn]);
        }

      const int key0 = k32 + ln, key1 = k32 + 16 + ln;
#pragma unroll
      for (int r = 0; r < 8; ++r) {
        const int qrow = q0 + mb + r;
        float s0 = (key0 <= qrow) ? s[0][r] : -1.0e30f;
        float s1 = (key1 <= qrow) ? s[1][r] : -1.0e30f;
        float mx = fmaxf(s0, s1);
        mx = fmaxf(mx, __shfl_xor(mx, 1, 32));
        mx = fmaxf(mx, __shfl_xor(mx, 2, 32));
        mx = fmaxf(mx, __shfl_xor(mx, 4, 32));
        mx = fmaxf(mx, __shfl_xor(mx, 8, 32));
        float mnew = fmaxf(mrow[r], mx);
        float corr = __expf(mrow[r] - mnew);
        mrow[r] = mnew;
        float p0 = __expf(s0 - mnew);
        float p1 = __expf(s1 - mnew);
        float ps = p0 + p1;
        ps += __shfl_xor(ps, 1, 32);
        ps += __shfl_xor(ps, 2, 32);
        ps += __shfl_xor(ps, 4, 32);
        ps += __shfl_xor(ps, 8, 32);
        lrow[r] = lrow[r] * corr + ps;
#pragma unroll
        for (int j = 0; j < 8; ++j) acc[j][r] *= corr;
        Pw[(mb + r) * 32 + ln]      = (_Float16)p0;
        Pw[(mb + r) * 32 + 16 + ln] = (_Float16)p1;
      }

      v16h pf = frag_a(Pw + ln * 32, hb);
#pragma unroll
      for (int j = 0; j < 8; ++j) {
        v16h vf = frag_b(Vs + (j * 16 + ln) * 32, hb);
        acc[j] = wmma16(pf, vf, acc[j]);
      }
    }
    __syncthreads();
  }

#pragma unroll
  for (int r = 0; r < 8; ++r) {
    float inv = 1.0f / lrow[r];
    int row = q0 + mb + r;
#pragma unroll
    for (int j = 0; j < 8; ++j)
      ctx[(size_t)row * HID + h * HD + j * 16 + ln] =
          (_Float16)(acc[j][r] * inv);
  }
}

// ---------------------------------------------------------------------------
// Launch
// ---------------------------------------------------------------------------
extern "C" void kernel_launch(void* const* d_in, const int* in_sizes, int n_in,
                              void* d_out, int out_size, void* d_ws, size_t ws_size,
                              hipStream_t stream) {
  const int*   positions = (const int*)d_in[0];
  const float* hidden    = (const float*)d_in[1];
  const float* w_qkv     = (const float*)d_in[2];
  const float* b_qkv     = (const float*)d_in[3];
  const float* w_dense   = (const float*)d_in[4];
  float* out = (float*)d_out;

  char* ws = (char*)d_ws;
  _Float16* hiddenH = (_Float16*)ws;  ws += (size_t)SEQ * HID * 2;
  _Float16* wqkvT   = (_Float16*)ws;  ws += (size_t)QKVD * HID * 2;  // N x K
  _Float16* wdenseT = (_Float16*)ws;  ws += (size_t)HID * HID * 2;   // N x K
  float*    qkv     = (float*)ws;     ws += (size_t)SEQ * QKVD * 4;
  _Float16* Qh      = (_Float16*)ws;  ws += (size_t)NH * SEQ * HD * 2;
  _Float16* Kh      = (_Float16*)ws;  ws += (size_t)KVG * SEQ * HD * 2;
  _Float16* Vt      = (_Float16*)ws;  ws += (size_t)KVG * HD * SEQ * 2;
  _Float16* ctx     = (_Float16*)ws;  ws += (size_t)SEQ * HID * 2;

  // 1) activations: plain cast; weights: transpose+cast (one-time)
  {
    int n4 = SEQ * HID / 4;
    cast_f16<<<(n4 + 255) / 256, 256, 0, stream>>>(hidden, hiddenH, n4);
  }
  transpose_cast<<<dim3(QKVD / 32, HID / 32), 256, 0, stream>>>(
      w_qkv, wqkvT, HID, QKVD);
  transpose_cast<<<dim3(HID / 32, HID / 32), 256, 0, stream>>>(
      w_dense, wdenseT, HID, HID);

  // 2) qkv = hidden @ w_qkv + b_qkv  (f32)
  gemm_f16<<<dim3(SEQ / 128, QKVD / 128), 256, 0, stream>>>(
      hiddenH, wqkvT, b_qkv, qkv, SEQ, QKVD, HID);

  // 3) rope + split + relayout
  rope_split<<<SEQ * 36, 128, 0, stream>>>(qkv, positions, Qh, Kh, Vt);

  // 4) causal flash attention -> ctx f16
  attn<<<dim3(SEQ / 128, NH), 256, 0, stream>>>(Qh, Kh, Vt, ctx);

  // 5) out = ctx @ w_dense  (f32)
  gemm_f16<<<dim3(SEQ / 128, HID / 128), 256, 0, stream>>>(
      ctx, wdenseT, nullptr, out, SEQ, HID, HID);

  (void)in_sizes; (void)n_in; (void)out_size; (void)ws_size;
}
